// Att_28879360099124
// MI455X (gfx1250) — matrix operand order repaired and checked by reading, not compiled, not run on hardware
//
#include <hip/hip_runtime.h>
#include <stdint.h>

// ---------------------------------------------------------------------------
// attW[b,s,t] = softmax_s( tanh( sum_h enc[s,b,h]*hid[t,b,h] ) )
// enc: [S,B,H] f32, hid: [TQ,B,H] f32, out: [B,S,TQ] f32
// tanh bounds scores to [-1,1] -> exp(tanh) in [1/e, e]: no max-trick needed.
// GEMM runs in bf16 WMMA (f32 accumulate): 137 GFLOP over ~576MB streamed
// needs multi-PFLOPS matmul to stay memory-bound -> v_wmma_f32_16x16x32_bf16.
// ---------------------------------------------------------------------------

#define TQn   1024
#define Bn    16
#define Hn    1024
#define Sn    4096

#define TILE_S 128
#define TILE_T 64
#define TILE_K 64
#define LDA    72   // padded LDS stride in bf16 elements (144B, 16B-aligned rows)

typedef __bf16        bf16x16 __attribute__((ext_vector_type(16)));
typedef __bf16        bf16x2  __attribute__((ext_vector_type(2)));
typedef float         f32x8   __attribute__((ext_vector_type(8)));
typedef unsigned int  u32x4   __attribute__((ext_vector_type(4)));

__device__ __forceinline__ uint32_t pk_bf16(float a, float b) {
#if __has_builtin(__builtin_amdgcn_cvt_pk_bf16_f32)
    bf16x2 p = __builtin_amdgcn_cvt_pk_bf16_f32(a, b);   // v_cvt_pk_bf16_f32
    return __builtin_bit_cast(uint32_t, p);
#else
    // fptrunc path: let the backend select the hardware bf16 convert
    bf16x2 p;
    p[0] = (__bf16)a;
    p[1] = (__bf16)b;
    return __builtin_bit_cast(uint32_t, p);
#endif
}

__global__ void att_zero_sums(float* __restrict__ sums) {
    int i = blockIdx.x * blockDim.x + threadIdx.x;
    if (i < Bn * TQn) sums[i] = 0.0f;
}

__global__ __launch_bounds__(256, 2)
void att_scores_kernel(const float* __restrict__ enc,
                       const float* __restrict__ hid,
                       float* __restrict__ out,
                       float* __restrict__ sums)
{
    __shared__ uint16_t sEnc[TILE_S * LDA];  // 18.4 KB
    __shared__ uint16_t sHid[TILE_T * LDA];  //  9.2 KB
    __shared__ float    colsum[TILE_T];

    const int b     = blockIdx.z;
    const int sTile = blockIdx.y * TILE_S;
    const int tTile = blockIdx.x * TILE_T;
    const int tid   = threadIdx.x;
    const int lane  = tid & 31;
    const int wave  = tid >> 5;
    const int sw    = wave & 3;   // 4 wave-rows along S (32 rows each)
    const int tw    = wave >> 2;  // 2 wave-cols along T (32 cols each)

    if (tid < TILE_T) colsum[tid] = 0.0f;

    f32x8 acc[2][2];
    for (int i = 0; i < 2; ++i)
        for (int j = 0; j < 2; ++j)
            for (int e = 0; e < 8; ++e) acc[i][j][e] = 0.0f;

    const float* encBase = enc + ((size_t)sTile * Bn + b) * Hn;
    const float* hidBase = hid + ((size_t)tTile * Bn + b) * Hn;
    const int lr = tid >> 4;          // 0..15: row within a load pass
    const int lc = (tid & 15) * 4;    // k offset (floats), float4 per thread

    for (int k0 = 0; k0 < Hn; k0 += TILE_K) {
        __syncthreads();
        // ---- stage enc tile [128 x 64] f32 -> bf16 LDS (8 passes) ----
        for (int p = 0; p < TILE_S; p += 16) {
            int row = p + lr;
            const float* src = encBase + (size_t)row * Bn * Hn + k0 + lc;
            __builtin_prefetch(src + TILE_K, 0, 3);   // speculative, next K-tile
            float4 v = *(const float4*)src;
            uint2 st; st.x = pk_bf16(v.x, v.y); st.y = pk_bf16(v.z, v.w);
            *(uint2*)(&sEnc[row * LDA + lc]) = st;
        }
        // ---- stage hid tile [64 x 64] f32 -> bf16 LDS (4 passes) ----
        for (int p = 0; p < TILE_T; p += 16) {
            int row = p + lr;
            const float* src = hidBase + (size_t)row * Bn * Hn + k0 + lc;
            float4 v = *(const float4*)src;
            uint2 st; st.x = pk_bf16(v.x, v.y); st.y = pk_bf16(v.z, v.w);
            *(uint2*)(&sHid[row * LDA + lc]) = st;
        }
        __syncthreads();

        // ---- 2 K-steps of 32, 2x2 subtiles per wave ----
        for (int ks = 0; ks < TILE_K; ks += 32) {
            bf16x16 afr[2], bfr[2];
            // A (16x32 bf16): lanes 0-15 M=0..15 K{0..7,16..23}; lanes 16-31 K{8..15,24..31}
            const int akoff = ks + ((lane >> 4) << 3);
            for (int ms = 0; ms < 2; ++ms) {
                const int r = sw * 32 + ms * 16 + (lane & 15);
                const uint16_t* pa = &sEnc[r * LDA + akoff];
                union { bf16x16 v; u32x4 q[2]; } ua;
                ua.q[0] = *(const u32x4*)pa;
                ua.q[1] = *(const u32x4*)(pa + 16);
                afr[ms] = ua.v;
            }
            // B (32x16 bf16): lane n holds column n; lanes 0-15 K{0..7,16..23}; lanes 16-31 K{8..15,24..31}
            const int bkoff = ks + ((lane >> 4) << 3);
            for (int ns = 0; ns < 2; ++ns) {
                const int c = tw * 32 + ns * 16 + (lane & 15);
                const uint16_t* pb = &sHid[c * LDA + bkoff];
                union { bf16x16 v; u32x4 q[2]; } ub;
                ub.q[0] = *(const u32x4*)pb;
                ub.q[1] = *(const u32x4*)(pb + 16);
                bfr[ns] = ub.v;
            }
            for (int ms = 0; ms < 2; ++ms)
                for (int ns = 0; ns < 2; ++ns)
                    acc[ms][ns] = __builtin_amdgcn_wmma_f32_16x16x32_bf16(
                        false, afr[ms], false, bfr[ns],
                        (short)0, acc[ms][ns], false, false);
        }
    }

    // ---- epilogue: e = exp(tanh(score)), store numerator, column sums ----
    for (int ms = 0; ms < 2; ++ms) {
        for (int ns = 0; ns < 2; ++ns) {
            const int srow0 = sTile + sw * 32 + ms * 16 + ((lane >> 4) << 3);
            const int tloc  = tw * 32 + ns * 16 + (lane & 15);
            const int tcol  = tTile + tloc;
            float psum = 0.0f;
            for (int i = 0; i < 8; ++i) {
                float x  = acc[ms][ns][i];
                float e2 = __expf(2.0f * x);                          // v_exp_f32
                float th = 1.0f - 2.0f * __builtin_amdgcn_rcpf(e2 + 1.0f);
                float e  = __expf(th);
                psum += e;
                out[((size_t)b * Sn + (srow0 + i)) * TQn + tcol] = e;
            }
            atomicAdd(&colsum[tloc], psum);                           // ds_add_f32
        }
    }
    __syncthreads();
    if (tid < TILE_T)
        atomicAdd(&sums[(size_t)b * TQn + tTile + tid], colsum[tid]); // global_atomic_add_f32
}

__global__ __launch_bounds__(256)
void att_norm_kernel(float* __restrict__ out, const float* __restrict__ sums)
{
    size_t idx = ((size_t)blockIdx.x * blockDim.x + threadIdx.x) * 4;
    size_t bb  = idx / ((size_t)Sn * TQn);
    int    t   = (int)(idx % TQn);
    float4 v  = *(float4*)(out + idx);
    const float4 sm = *(const float4*)(sums + bb * TQn + t);
    v.x *= __builtin_amdgcn_rcpf(sm.x);
    v.y *= __builtin_amdgcn_rcpf(sm.y);
    v.z *= __builtin_amdgcn_rcpf(sm.z);
    v.w *= __builtin_amdgcn_rcpf(sm.w);
    *(float4*)(out + idx) = v;
}

extern "C" void kernel_launch(void* const* d_in, const int* in_sizes, int n_in,
                              void* d_out, int out_size, void* d_ws, size_t ws_size,
                              hipStream_t stream) {
    const float* hid = (const float*)d_in[0];   // hiddenState [TQ,B,H]
    const float* enc = (const float*)d_in[1];   // encoderOut  [S,B,H]
    float* out  = (float*)d_out;                // [B,S,TQ]
    float* sums = (float*)d_ws;                 // B*TQ f32 = 64 KB

    att_zero_sums<<<(Bn * TQn + 255) / 256, 256, 0, stream>>>(sums);

    dim3 grid(TQn / TILE_T, Sn / TILE_S, Bn);   // 16 x 32 x 16
    att_scores_kernel<<<grid, 256, 0, stream>>>(enc, hid, out, sums);

    const size_t total = (size_t)Bn * Sn * TQn;
    att_norm_kernel<<<(unsigned)(total / 4 / 256), 256, 0, stream>>>(out, sums);
}
